// CorrLSTM_flow_29609504538951
// MI455X (gfx1250) — matrix operand order repaired and checked by reading, not compiled, not run on hardware
//
#include <hip/hip_runtime.h>
#include <hip/hip_bf16.h>
#include <math.h>

typedef __attribute__((ext_vector_type(8)))  __bf16 v8bf;
typedef __attribute__((ext_vector_type(16))) __bf16 v16bf;
typedef __attribute__((ext_vector_type(8)))  float  v8f;

__device__ __forceinline__ __bf16 f2bf(float f) {
    unsigned u = __builtin_bit_cast(unsigned, f);
    unsigned r = u + 0x7fffu + ((u >> 16) & 1u);   // round-to-nearest-even
    unsigned short s = (unsigned short)(r >> 16);
    return __builtin_bit_cast(__bf16, s);
}

// ---------------------------------------------------------------------------
// f32 -> bf16 copy into a channel-offset region of a [B][dstC][HW] bf16 tensor
// ---------------------------------------------------------------------------
__global__ void k_copy_bf16(const float* __restrict__ src, __bf16* __restrict__ dst,
                            int Bn, int Cs, int HW, int dstC, int c_off) {
    long long i = (long long)blockIdx.x * blockDim.x + threadIdx.x;
    long long tot = (long long)Bn * Cs * HW;
    if (i >= tot) return;
    int p = (int)(i % HW);
    int c = (int)((i / HW) % Cs);
    int b = (int)(i / ((long long)HW * Cs));
    dst[((long long)b * dstC + c_off + c) * HW + p] = f2bf(src[i]);
}

// ---------------------------------------------------------------------------
// OIHW f32 weights -> zero-padded [Mpad][Kpad] bf16, K = (ky*3+kx)*Cin + ci
// ---------------------------------------------------------------------------
__global__ void k_pack_w(const float* __restrict__ w, __bf16* __restrict__ dst,
                         int M, int Cin, int Mpad, int Kpad) {
    long long i = (long long)blockIdx.x * blockDim.x + threadIdx.x;
    long long tot = (long long)Mpad * Kpad;
    if (i >= tot) return;
    int k = (int)(i % Kpad);
    int m = (int)(i / Kpad);
    float v = 0.f;
    int Kreal = 9 * Cin;
    if (m < M && k < Kreal) {
        int j = k / Cin;           // ky*3+kx
        int ci = k - j * Cin;
        v = w[((long long)m * Cin + ci) * 9 + j];
    }
    dst[i] = f2bf(v);
}

// ---------------------------------------------------------------------------
// 7x7 spatial correlation + leaky ReLU.
// block = one (b, y) row. LDS stages channel chunks of f1 row and f2 halo.
// Writes f32 to d_out corr section and bf16 into lstm_in concat buffer.
// ---------------------------------------------------------------------------
__global__ __launch_bounds__(256)
void k_corr(const float* __restrict__ f1, const float* __restrict__ f2,
            float* __restrict__ corr_out, __bf16* __restrict__ lstm_in,
            int C, int H, int W, int chunk, int C2, int c_off) {
    extern __shared__ float sm[];
    float* f1c = sm;                         // chunk*W
    float* f2c = f1c + chunk * W;            // chunk*7*(W+6)
    float* red = f2c + chunk * 7 * (W + 6);  // 49*W
    const int b = blockIdx.x / H;
    const int y = blockIdx.x % H;
    const int tid = threadIdx.x;
    const int x = tid % W;
    const int q = tid / W;
    const int NPH = blockDim.x / W;
    const int cpp = chunk / NPH;
    const int W6 = W + 6;

    float acc[49];
#pragma unroll
    for (int d = 0; d < 49; ++d) acc[d] = 0.f;
    for (int i = tid; i < 49 * W; i += blockDim.x) red[i] = 0.f;

    for (int cc = 0; cc < C; cc += chunk) {
        __syncthreads();
        for (int i = tid; i < chunk * W; i += blockDim.x) {
            int cl = i / W, xx = i % W;
            f1c[i] = f1[(((long long)b * C + cc + cl) * H + y) * W + xx];
        }
        for (int i = tid; i < chunk * 7 * W6; i += blockDim.x) {
            int cl = i / (7 * W6);
            int rem = i % (7 * W6);
            int dy = rem / W6;
            int xx = rem % W6 - 3;
            int yy = y + dy - 3;
            float v = 0.f;
            if (yy >= 0 && yy < H && xx >= 0 && xx < W)
                v = f2[(((long long)b * C + cc + cl) * H + yy) * W + xx];
            f2c[i] = v;
        }
        __syncthreads();
        for (int cl = q * cpp; cl < (q + 1) * cpp; ++cl) {
            float a = f1c[cl * W + x];
            const float* row = &f2c[cl * 7 * W6];
#pragma unroll
            for (int di = 0; di < 7; ++di)
#pragma unroll
                for (int dj = 0; dj < 7; ++dj)
                    acc[di * 7 + dj] += a * row[di * W6 + x + dj];
        }
    }
    __syncthreads();
#pragma unroll
    for (int d = 0; d < 49; ++d) atomicAdd(&red[d * W + x], acc[d]);
    __syncthreads();
    if (q == 0) {
        for (int d = 0; d < 49; ++d) {
            float v = red[d * W + x];
            v = v > 0.f ? v : 0.01f * v;
            corr_out[(((long long)b * 49 + d) * H + y) * W + x] = v;
            lstm_in[(((long long)b * C2 + c_off + d) * H + y) * W + x] = f2bf(v);
        }
    }
}

// ---------------------------------------------------------------------------
// Implicit-GEMM conv3x3 via v_wmma_f32_16x16x32_bf16.
// D[M=Cout][N=B*H*W] = sum_K Wp[M][K] * im2col(Bin)[K][N], K=(ky*3+kx)*Cin+ci
// Block: 64(M) x 128(N), 8 waves each computing a 32x32 tile (2x2 WMMA accs).
// relu_mode=1: bias+ReLU -> bf16 (summary into lstm_in); 0: bias -> f32 (cx).
// ---------------------------------------------------------------------------
#define MT 64
#define NT 128
#define KT 32
#define ASTR 48   // bf16 elems per A row (96B, 16B-aligned b128 frag loads)
#define BSTR 48   // bf16 elems per B row (B stored [n][k])

__global__ __launch_bounds__(256)
void k_conv_gemm(const __bf16* __restrict__ Bin, const __bf16* __restrict__ Wp,
                 const float* __restrict__ bias,
                 float* __restrict__ outF, __bf16* __restrict__ outB,
                 int Cin, int M, int Kpad, int H, int W, int Bn,
                 int dstC, int relu_mode) {
    __shared__ __align__(16) __bf16 Asub[MT * ASTR];
    __shared__ __align__(16) __bf16 Bsub[NT * BSTR];
    const int HW = H * W;
    const int N = Bn * HW;
    const int Kreal = 9 * Cin;
    const int n0 = blockIdx.x * NT;
    const int m0 = blockIdx.y * MT;
    const int tid = threadIdx.x;
    const int lane = tid & 31;
    const int wid = tid >> 5;
    const int wm = wid >> 2;    // 0..1 -> 32-row half
    const int wn = wid & 3;     // 0..3 -> 32-col quarter
    const int lm = lane & 15;
    const int kh = lane >> 4;

    v8f acc[2][2];
    {
        v8f z;
#pragma unroll
        for (int e = 0; e < 8; ++e) z[e] = 0.f;
        acc[0][0] = z; acc[0][1] = z; acc[1][0] = z; acc[1][1] = z;
    }

    const int a_m = tid >> 2;           // 0..63
    const int a_k = (tid & 3) * 8;      // 0,8,16,24
    const int b_k = tid >> 3;           // 0..31
    const int b_n = tid & 7;            // 0..7 (+8*i)

    for (int k0 = 0; k0 < Kpad; k0 += KT) {
        // --- A slab: one 16B global load -> LDS ---
        v8bf av = *(const v8bf*)(Wp + (size_t)(m0 + a_m) * Kpad + k0 + a_k);
        *(v8bf*)(Asub + a_m * ASTR + a_k) = av;
        if (k0 + KT < Kpad)
            __builtin_prefetch(Wp + (size_t)(m0 + a_m) * Kpad + k0 + KT + a_k, 0, 0);

        // --- B slab: im2col gather, one K row per thread ---
        int k = k0 + b_k;
        int j = 0, ci = 0, dy = 0, dx = 0;
        bool kval = k < Kreal;
        if (kval) {
            j = k / Cin; ci = k - j * Cin;
            dy = j / 3 - 1; dx = j - (j / 3) * 3 - 1;
        }
#pragma unroll 4
        for (int i = 0; i < 16; ++i) {
            int n = n0 + b_n + i * 8;
            __bf16 v = __builtin_bit_cast(__bf16, (unsigned short)0);
            if (kval && n < N) {
                int bimg = n / HW;
                int rem = n - bimg * HW;
                int yy = rem / W + dy;
                int xx = rem - (rem / W) * W + dx;
                if (yy >= 0 && yy < H && xx >= 0 && xx < W)
                    v = Bin[((size_t)(bimg * Cin + ci) * H + yy) * W + xx];
            }
            Bsub[(b_n + i * 8) * BSTR + b_k] = v;
        }
        __syncthreads();

        // --- fragments (ISA 7.12.2 layouts) + 4 WMMAs per wave ---
#pragma unroll
        for (int mi = 0; mi < 2; ++mi) {
            union { v16bf v; v8bf h[2]; } af;
            const __bf16* ab = Asub + (wm * 32 + mi * 16 + lm) * ASTR;
            af.h[0] = *(const v8bf*)(ab + kh * 8);        // K {0..7} / {8..15}
            af.h[1] = *(const v8bf*)(ab + 16 + kh * 8);   // K {16..23} / {24..31}
#pragma unroll
            for (int ni = 0; ni < 2; ++ni) {
                union { v16bf v; v8bf h[2]; } bfv;
                const __bf16* bb = Bsub + (wn * 32 + ni * 16 + lm) * BSTR + kh * 8;
                bfv.h[0] = *(const v8bf*)(bb);
                bfv.h[1] = *(const v8bf*)(bb + 16);
                acc[mi][ni] = __builtin_amdgcn_wmma_f32_16x16x32_bf16(
                    false, af.v, false, bfv.v, (short)0, acc[mi][ni], false, false);
            }
        }
        __syncthreads();
    }

    // --- epilogue: C/D layout M = r + 8*kh, N = lm ---
#pragma unroll
    for (int mi = 0; mi < 2; ++mi)
#pragma unroll
        for (int ni = 0; ni < 2; ++ni)
#pragma unroll
            for (int r = 0; r < 8; ++r) {
                int m = m0 + wm * 32 + mi * 16 + r + 8 * kh;
                int n = n0 + wn * 32 + ni * 16 + lm;
                if (m < M && n < N) {
                    float v = acc[mi][ni][r] + bias[m];
                    int bimg = n / HW;
                    int rem = n - bimg * HW;
                    size_t di = ((size_t)bimg * dstC + m) * HW + rem;
                    if (relu_mode) { v = v > 0.f ? v : 0.f; outB[di] = f2bf(v); }
                    else           { outF[di] = v; }
                }
            }
}

// ---------------------------------------------------------------------------
// LSTM gate fusion (fp32).
// ---------------------------------------------------------------------------
__global__ void k_gate(const float* __restrict__ cx, const float* __restrict__ cprev,
                       float* __restrict__ hout, float* __restrict__ cout_,
                       int ch, int HW, int Bn) {
    long long i = (long long)blockIdx.x * blockDim.x + threadIdx.x;
    long long tot = (long long)Bn * ch * HW;
    if (i >= tot) return;
    int p = (int)(i % HW);
    int m = (int)((i / HW) % ch);
    int b = (int)(i / ((long long)HW * ch));
    size_t base = (size_t)b * 4 * ch * HW;
    float ig = cx[base + (size_t)m * HW + p];
    float fg = cx[base + (size_t)(m + ch) * HW + p];
    float og = cx[base + (size_t)(m + 2 * ch) * HW + p];
    float gg = cx[base + (size_t)(m + 3 * ch) * HW + p];
    ig = 1.f / (1.f + __expf(-ig));
    fg = 1.f / (1.f + __expf(-fg));
    og = 1.f / (1.f + __expf(-og));
    gg = tanhf(gg);
    float cn = fg * cprev[i] + ig * gg;
    cout_[i] = cn;
    hout[i] = og * tanhf(cn);
}

// ---------------------------------------------------------------------------
extern "C" void kernel_launch(void* const* d_in, const int* in_sizes, int n_in,
                              void* d_out, int out_size, void* d_ws, size_t ws_size,
                              hipStream_t stream) {
    (void)in_sizes; (void)n_in; (void)out_size; (void)ws_size;
    const int Bn = 8;
    const int CINa[6] = {512, 1024, 512, 256, 256, 256};
    const int Sa[6]   = {64, 32, 16, 8, 4, 2};
    int cha[6];
    for (int i = 0; i < 6; ++i) cha[i] = 49 + CINa[i] / 8;

    size_t ho[6], co[6], ro[6], off = 0;
    for (int i = 0; i < 6; ++i) { ho[i] = off; off += (size_t)Bn * cha[i] * Sa[i] * Sa[i]; }
    for (int i = 0; i < 6; ++i) { co[i] = off; off += (size_t)Bn * cha[i] * Sa[i] * Sa[i]; }
    for (int i = 0; i < 6; ++i) { ro[i] = off; off += (size_t)Bn * 49 * Sa[i] * Sa[i]; }
    float* out = (float*)d_out;

    char* ws = (char*)d_ws;
    size_t woff = 0;
    auto alloc = [&](size_t bytes) -> size_t {
        size_t o = woff; woff += (bytes + 255) & ~(size_t)255; return o;
    };

    for (int i = 0; i < 6; ++i) {
        const int C = CINa[i], s = Sa[i], H = s, W = s, HW = s * s, ch = cha[i];
        const int C2 = 2 * ch, Ms = C / 8, Ml = 4 * ch;
        const int KpS = ((9 * C + 31) / 32) * 32;
        const int KpL = ((9 * C2 + 31) / 32) * 32;
        const int MpS = ((Ms + 63) / 64) * 64;
        const int MpL = ((Ml + 63) / 64) * 64;

        const float* x  = (const float*)d_in[8 * i + 0];
        const float* xp = (const float*)d_in[8 * i + 1];
        const float* h  = (const float*)d_in[8 * i + 2];
        const float* c  = (const float*)d_in[8 * i + 3];
        const float* sw = (const float*)d_in[8 * i + 4];
        const float* sb = (const float*)d_in[8 * i + 5];
        const float* lw = (const float*)d_in[8 * i + 6];
        const float* lb = (const float*)d_in[8 * i + 7];

        __bf16* xb  = (__bf16*)(ws + alloc((size_t)Bn * C * HW * 2));
        __bf16* lin = (__bf16*)(ws + alloc((size_t)Bn * C2 * HW * 2));
        __bf16* wps = (__bf16*)(ws + alloc((size_t)MpS * KpS * 2));
        __bf16* wpl = (__bf16*)(ws + alloc((size_t)MpL * KpL * 2));
        float*  cx  = (float*)(ws + alloc((size_t)Bn * Ml * HW * 4));

        { // x -> bf16
            long long tot = (long long)Bn * C * HW;
            k_copy_bf16<<<(unsigned)((tot + 255) / 256), 256, 0, stream>>>(
                x, xb, Bn, C, HW, C, 0);
        }
        { // h -> lstm_in channels [ch, 2ch)
            long long tot = (long long)Bn * ch * HW;
            k_copy_bf16<<<(unsigned)((tot + 255) / 256), 256, 0, stream>>>(
                h, lin, Bn, ch, HW, C2, ch);
        }
        { // pack weights
            long long tot = (long long)MpS * KpS;
            k_pack_w<<<(unsigned)((tot + 255) / 256), 256, 0, stream>>>(sw, wps, Ms, C, MpS, KpS);
            tot = (long long)MpL * KpL;
            k_pack_w<<<(unsigned)((tot + 255) / 256), 256, 0, stream>>>(lw, wpl, Ml, C2, MpL, KpL);
        }
        { // correlation -> d_out + lstm_in channels [Ms, Ms+49)
            int NPH = 256 / W;
            int chunk = NPH > 32 ? NPH : 32;
            size_t sh = ((size_t)chunk * W + (size_t)chunk * 7 * (W + 6) + (size_t)49 * W) * 4;
            k_corr<<<Bn * H, 256, sh, stream>>>(x, xp, out + ro[i], lin, C, H, W, chunk, C2, Ms);
        }
        { // summary conv (WMMA) -> lstm_in channels [0, Ms), ReLU, bf16
            dim3 g((unsigned)((Bn * HW + NT - 1) / NT), (unsigned)((Ms + MT - 1) / MT));
            k_conv_gemm<<<g, 256, 0, stream>>>(xb, wps, sb, nullptr, lin,
                                               C, Ms, KpS, H, W, Bn, C2, 1);
        }
        { // LSTM conv (WMMA) -> cx f32
            dim3 g((unsigned)((Bn * HW + NT - 1) / NT), (unsigned)((Ml + MT - 1) / MT));
            k_conv_gemm<<<g, 256, 0, stream>>>(lin, wpl, lb, cx, nullptr,
                                               C2, Ml, KpL, H, W, Bn, Ml, 0);
        }
        { // gates -> d_out h/c
            long long tot = (long long)Bn * ch * HW;
            k_gate<<<(unsigned)((tot + 255) / 256), 256, 0, stream>>>(
                cx, c, out + ho[i], out + co[i], ch, HW, Bn);
        }
    }
}
